// RV2BEVFrustumAttn_59614146068887
// MI455X (gfx1250) — matrix operand order repaired and checked by reading, not compiled, not run on hardware
//
#include <hip/hip_runtime.h>

// ---------------------------------------------------------------------------
// Types for CDNA5 WMMA (wave32): 16x16x32 bf16 -> f32 accumulate
// ---------------------------------------------------------------------------
typedef __attribute__((ext_vector_type(16))) __bf16 v16bf;
typedef __attribute__((ext_vector_type(8)))  __bf16 v8bf;
typedef __attribute__((ext_vector_type(8)))  float  v8f;
typedef __attribute__((ext_vector_type(4)))  float  v4f;

#define DEV __device__ __forceinline__

// Problem constants
#define BATCH 16
#define DMODEL 128
#define NQ_ 20
#define NH_ 8
#define NP_ 6
#define HRV_ 2
#define WRV_ 64
#define CRV_ 256
#define CBEV_ 256
#define HBEV_ 128
#define WBEV_ 128
#define PBEV (HBEV_*WBEV_)        // 16384
#define MRV (BATCH*HRV_*WRV_)     // 2048
#define LQ  (HRV_*WRV_*NQ_)       // 2560

__constant__ float c_elev[32] = {
  -30.67f, -29.33f, -28.0f, -26.66f, -25.33f, -24.0f, -22.67f, -21.33f,
  -20.0f, -18.67f, -17.33f, -16.0f, -14.67f, -13.33f, -12.0f, -10.67f,
  -9.33f, -8.0f, -6.66f, -5.33f, -4.0f, -2.67f, -1.33f, 0.0f,
  1.33f, 2.67f, 4.0f, 5.33f, 6.67f, 8.0f, 9.33f, 10.67f };

DEV float gelu_f(float x){ return 0.5f*x*(1.0f + erff(x*0.7071067811865475f)); }

DEV v8bf cvt8(const float* p){
  v4f a = *(const v4f*)p;
  v4f b = *(const v4f*)(p + 4);
  v8bf r;
  r[0]=(__bf16)a[0]; r[1]=(__bf16)a[1]; r[2]=(__bf16)a[2]; r[3]=(__bf16)a[3];
  r[4]=(__bf16)b[0]; r[5]=(__bf16)b[1]; r[6]=(__bf16)b[2]; r[7]=(__bf16)b[3];
  return r;
}

// ---------------------------------------------------------------------------
// Generic WMMA GEMM, one 16 x (16*NT) strip per wave:
//   C[M,N] = epi( A[M,K] * W[N,K]^T + bias[N] )
//   A: f32 (ABF=false) or bf16 (ABF=true), row-major (M,K)
//   W: bf16 row-major (N,K)  -> lane-contiguous K loads for B fragment
// The A fragment is loaded once per K-step and reused by NT WMMAs (cuts
// streaming-A traffic NT-fold); all NT B fragments are loaded before the
// WMMA burst so the loads clause together and overlap WMMA issue.
// EPI: 0 bias, 1 bias+gelu, 2 bias + extra[(m%emod)*N+n]  (positional-enc add)
//      3 bias + extra[(m/emod)*N+n], scatter write C[(m/emod)*N*emod + n*emod + m%emod]
//      4 bias + extra[(m/emod)*N+n], normal write
// 8 waves (256 threads) per block.
// ---------------------------------------------------------------------------
template<int EPI, bool ABF, int NT>
__global__ void gemm_wmma(const void* __restrict__ Ap, const __bf16* __restrict__ W,
                          const float* __restrict__ bias, float* __restrict__ C,
                          __bf16* __restrict__ Cb, const float* __restrict__ extra,
                          int emod, int M, int N, int K){
  const int wave = threadIdx.x >> 5;
  const int lane = threadIdx.x & 31;
  const int strips = (N >> 4) / NT;
  const int strip = blockIdx.x * 8 + wave;
  if (strip >= (M >> 4) * strips) return;           // uniform per wave: EXEC stays full
  const int tm  = strip / strips;
  const int tn0 = (strip % strips) * NT;
  const int arow = tm*16 + (lane & 15);
  const int ncol = (lane & 15);
  const int hi   = (lane < 16) ? 0 : 1;

  v8f acc[NT];
  #pragma unroll
  for (int nt = 0; nt < NT; ++nt)
    #pragma unroll
    for (int r = 0; r < 8; ++r) acc[nt][r] = 0.f;

  const float*  Af = (const float*)Ap;
  const __bf16* Ab = (const __bf16*)Ap;

  for (int k0 = 0; k0 < K; k0 += 32){
    // A fragment: lanes 0-15 -> K k0..7 / k0+16..23 ; lanes 16-31 -> K k0+8..15 / k0+24..31
    v8bf alo, ahi;
    const int ka = k0 + hi*8;
    if (ABF){
      const __bf16* ap = Ab + (size_t)arow*K + ka;
      alo = *(const v8bf*)ap;
      ahi = *(const v8bf*)(ap + 16);
      if (k0 + 32 < K) __builtin_prefetch(ap + 32, 0, 3);   // global_prefetch_b8 (near)
    } else {
      const float* ap = Af + (size_t)arow*K + ka;
      alo = cvt8(ap);
      ahi = cvt8(ap + 16);
      if (k0 + 32 < K) __builtin_prefetch(ap + 32, 0, 3);
    }
    v16bf afrag = __builtin_shufflevector(alo, ahi, 0,1,2,3,4,5,6,7,8,9,10,11,12,13,14,15);

    // B fragments: lanes 0-15 -> N=lane, K k0..7 / k0+16..23 ; lanes 16-31 -> N=lane-16, K k0+8..15 / k0+24..31
    // Phase 1: issue all NT fragment loads (clause-able), Phase 2: WMMA burst.
    v16bf bfrag[NT];
    #pragma unroll
    for (int nt = 0; nt < NT; ++nt){
      const int nn = (tn0 + nt)*16 + ncol;
      const __bf16* wp = W + (size_t)nn*K + k0 + hi*8;
      v8bf blo = *(const v8bf*)wp;
      v8bf bhi = *(const v8bf*)(wp + 16);
      bfrag[nt] = __builtin_shufflevector(blo, bhi, 0,1,2,3,4,5,6,7,8,9,10,11,12,13,14,15);
    }
    #pragma unroll
    for (int nt = 0; nt < NT; ++nt)
      acc[nt] = __builtin_amdgcn_wmma_f32_16x16x32_bf16(false, afrag, false, bfrag[nt],
                                                        (short)0, acc[nt], false, false);
  }

  // C/D layout: VGPR r, lanes 0-15 -> M=r, N=lane ; lanes 16-31 -> M=8+r, N=lane-16
  const int mbase = tm*16 + ((lane < 16) ? 0 : 8);
  #pragma unroll
  for (int nt = 0; nt < NT; ++nt){
    const int nn = (tn0 + nt)*16 + ncol;
    const float bn = bias ? bias[nn] : 0.f;
    #pragma unroll
    for (int r = 0; r < 8; ++r){
      const int m = mbase + r;
      float v = acc[nt][r] + bn;
      if (EPI == 1) v = gelu_f(v);
      if (EPI == 2) v += extra[(size_t)(m % emod)*N + nn];
      if (EPI == 3 || EPI == 4) v += extra[(size_t)(m / emod)*N + nn];
      if (EPI == 3){
        C[(size_t)(m / emod)*N*emod + (size_t)nn*emod + (m % emod)] = v;
      } else {
        if (C)  C[(size_t)m*N + nn] = v;
        if (Cb) Cb[(size_t)m*N + nn] = (__bf16)v;
      }
    }
  }
}

// ---------------------------------------------------------------------------
// bev (B,C,P) f32 -> bevT (B,P,C) bf16 via 32x32 LDS tile (coalesced both ways)
// ---------------------------------------------------------------------------
__global__ void bev_transpose(const float* __restrict__ bev, __bf16* __restrict__ bevT){
  __shared__ float tile[32][33];
  int blk = blockIdx.x;
  const int pblk = blk % (PBEV/32); blk /= (PBEV/32);
  const int cblk = blk % (CBEV_/32);
  const int b    = blk / (CBEV_/32);
  const int t = threadIdx.x;
  const float* src = bev + ((size_t)b*CBEV_ + cblk*32)*PBEV + pblk*32;
  #pragma unroll
  for (int i = 0; i < 4; ++i){
    const int idx = t + 256*i;
    tile[idx >> 5][idx & 31] = src[(size_t)(idx >> 5)*PBEV + (idx & 31)];
  }
  __syncthreads();
  __bf16* dst = bevT + ((size_t)b*PBEV + pblk*32)*CBEV_ + cblk*32;
  #pragma unroll
  for (int i = 0; i < 4; ++i){
    const int idx = t + 256*i;
    dst[(size_t)(idx >> 5)*CBEV_ + (idx & 31)] = (__bf16)tile[idx & 31][idx >> 5];
  }
}

// ---------------------------------------------------------------------------
// Frustum geometry: UX/UY/UZ (s,q) and normalized mean direction UD (3,s)
// ---------------------------------------------------------------------------
__global__ void geom_kernel(float* UX, float* UY, float* UZ, float* UD){
  const int s = threadIdx.x; if (s >= 128) return;
  const int h = s >> 6, w = s & 63;
  const float PI = 3.14159265358979323846f;
  const float az_step = 2.f*PI/64.f;
  const float az0 = -PI + w*az_step + 0.5f*az_step;
  float mx = 0.f, my = 0.f, mz = 0.f;
  for (int iw = 0; iw < 5; ++iw){
    const float az = az0 + ((iw + 0.5f)/5.f - 0.5f)*az_step;
    const float ca = cosf(az), sa = sinf(az);
    for (int q = 0; q < 4; ++q){
      const float e = c_elev[31 - (h*16 + 4*q)] * (PI/180.f);
      const float ce = cosf(e), se = sinf(e);
      const int n = iw*4 + q;
      const float ux = ca*ce, uy = sa*ce, uz = se;
      UX[s*NQ_+n] = ux; UY[s*NQ_+n] = uy; UZ[s*NQ_+n] = uz;
      mx += ux; my += uy; mz += uz;
    }
  }
  mx *= (1.f/20.f); my *= (1.f/20.f); mz *= (1.f/20.f);
  const float inv = rsqrtf(mx*mx + my*my + mz*mz);
  UD[0*128+s] = mx*inv; UD[1*128+s] = my*inv; UD[2*128+s] = mz*inv;
}

// BEV positional encoding, transposed: PEt[p][d]
__global__ void pe_kernel(float* __restrict__ PEt){
  const int i = blockIdx.x*256 + threadIdx.x;
  if (i >= PBEV*DMODEL) return;
  const int d = i & 127, p = i >> 7;
  const int x = p & 127, y = p >> 7;
  const int k = (d & 63) >> 1;
  const float div = expf(-(float)(2*k) * (9.210340371976184f/64.f)); // ln(1e4)/64
  const float arg = ((d < 64) ? (float)x : (float)y) * div;
  PEt[i] = (d & 1) ? cosf(arg) : sinf(arg);
}

// biasv[d] = mvw[d,:] . bv + mvb[d]
__global__ void biasv_kernel(const float* __restrict__ mvw, const float* __restrict__ bv,
                             const float* __restrict__ mvb, float* __restrict__ biasv){
  const int d = threadIdx.x; if (d >= 128) return;
  float a = mvb[d];
  for (int e = 0; e < 128; ++e) a += mvw[d*128 + e]*bv[e];
  biasv[d] = a;
}

__global__ void cvt_bf16_kernel(const float* __restrict__ s, __bf16* __restrict__ d, int n){
  const int i = blockIdx.x*256 + threadIdx.x;
  if (i < n) d[i] = (__bf16)s[i];
}
__global__ void cvt_t_bf16_kernel(const float* __restrict__ s, __bf16* __restrict__ d, int R, int Cc){
  const int i = blockIdx.x*256 + threadIdx.x;
  if (i < R*Cc){ const int r = i / Cc, c = i % Cc; d[(size_t)c*R + r] = (__bf16)s[i]; }
}
__global__ void copy_f32_kernel(const float* __restrict__ s, float* __restrict__ d, int n){
  const int i = blockIdx.x*256 + threadIdx.x;
  if (i < n) d[i] = s[i];
}

// rin = [x (256ch from x_rv NHWC), U_DIR (3ch)]; out[b][c][s] = rw1 . rin + rb1
__global__ void rin_conv_kernel(const float* __restrict__ x_rv, const float* __restrict__ UD,
                                const float* __restrict__ rw1, const float* __restrict__ rb1,
                                float* __restrict__ out){
  const int idx = blockIdx.x*256 + threadIdx.x;
  if (idx >= BATCH*128*128) return;
  const int s = idx & 127, c = (idx >> 7) & 127, b = idx >> 14;
  const float* xr = x_rv + ((size_t)b*128 + s)*CRV_;
  const float* wr = rw1 + c*259;
  float acc = rb1[c];
  for (int k = 0; k < 256; ++k) acc += wr[k]*xr[k];
  for (int k = 0; k < 3;   ++k) acc += wr[256+k]*UD[k*128 + s];
  out[idx] = acc;
}

// in-place groupnorm(+gelu) over (Cg channels x 128 spatial) per (b,group)
__global__ void gn_gelu_kernel(float* __restrict__ x, const float* __restrict__ g,
                               const float* __restrict__ bt, int C, int Cg){
  __shared__ float s1[256], s2[256];
  const int G = C / Cg;
  const int b = blockIdx.x / G, grp = blockIdx.x % G;
  float* base = x + ((size_t)b*C + grp*Cg)*128;
  const int n = Cg*128;
  const int t = threadIdx.x;
  float a = 0.f, a2 = 0.f;
  for (int i = t; i < n; i += 256){ const float v = base[i]; a += v; a2 += v*v; }
  s1[t] = a; s2[t] = a2; __syncthreads();
  for (int o = 128; o > 0; o >>= 1){
    if (t < o){ s1[t] += s1[t+o]; s2[t] += s2[t+o]; } __syncthreads();
  }
  const float mean = s1[0]/n;
  const float inv  = rsqrtf(s2[0]/n - mean*mean + 1e-5f);
  for (int i = t; i < n; i += 256){
    const int c = grp*Cg + (i >> 7);
    base[i] = gelu_f((base[i]-mean)*inv*g[c] + bt[c]);
  }
}

// circular-pad 3x3 conv: in (B,128,2,64) -> out (B,64,2,64)
__global__ void conv3x3_kernel(const float* __restrict__ in, const float* __restrict__ w,
                               float* __restrict__ out){
  const int idx = blockIdx.x*256 + threadIdx.x;
  if (idx >= BATCH*64*128) return;
  const int s = idx & 127, co = (idx >> 7) & 63, b = idx >> 13;
  const int h = s >> 6, wd = s & 63;
  const float* inb = in + (size_t)b*128*128;
  const float* wc = w + (size_t)co*128*9;
  float acc = 0.f;
  for (int ci = 0; ci < 128; ++ci){
    const float* ip = inb + ci*128;
    const float* wp = wc + ci*9;
    #pragma unroll
    for (int dh = 0; dh < 3; ++dh){
      const int hh = (h + dh + 1) & 1;
      #pragma unroll
      for (int dw = 0; dw < 3; ++dw){
        const int ww = (wd + dw + 63) & 63;
        acc += wp[dh*3+dw]*ip[hh*64 + ww];
      }
    }
  }
  out[idx] = acc;
}

// depth head + frustum unprojection -> ref points (B, LQ, 2)
__global__ void depth_ref_kernel(const float* __restrict__ h2, const float* __restrict__ rw3,
                                 const float* __restrict__ rb3, const float* __restrict__ UX,
                                 const float* __restrict__ UY, const float* __restrict__ UZ,
                                 const float* __restrict__ l2e, float* __restrict__ ref){
  const int idx = blockIdx.x*256 + threadIdx.x;
  if (idx >= BATCH*128) return;
  const int s = idx & 127, b = idx >> 7;
  const float* hb = h2 + (size_t)b*64*128 + s;
  const float* L = l2e + b*16;
  for (int q = 0; q < NQ_; ++q){
    float lg = rb3[q];
    const float* wq_ = rw3 + q*64;
    for (int ci = 0; ci < 64; ++ci) lg += wq_[ci]*hb[ci*128];
    const float mu = 51.2f / (1.f + expf(-lg));
    const float xl = mu*UX[s*NQ_+q], yl = mu*UY[s*NQ_+q], zl = mu*UZ[s*NQ_+q];
    const float px = xl*L[0] + yl*L[4] + zl*L[8]  + L[12];
    const float py = xl*L[1] + yl*L[5] + zl*L[9]  + L[13];
    const float rx = fminf(fmaxf((px + 51.2f)*(1.f/102.4f), 0.f), 1.f);
    const float ry = fminf(fmaxf((py + 51.2f)*(1.f/102.4f), 0.f), 1.f);
    const size_t o = ((size_t)b*LQ + s*NQ_ + q)*2;
    ref[o] = rx; ref[o+1] = ry;
  }
}

// softmax over NP=6 attention logits per (row, head); logits live in OA[:,96:144]
__global__ void softmax_aw_kernel(const float* __restrict__ OA, float* __restrict__ AW){
  const int i = blockIdx.x*256 + threadIdx.x;
  if (i >= MRV*NH_) return;
  const int m = i >> 3, h = i & 7;
  const float* lg = OA + (size_t)m*144 + 96 + h*NP_;
  float mx = lg[0];
  #pragma unroll
  for (int p = 1; p < NP_; ++p) mx = fmaxf(mx, lg[p]);
  float e[NP_], sum = 0.f;
  #pragma unroll
  for (int p = 0; p < NP_; ++p){ e[p] = expf(lg[p]-mx); sum += e[p]; }
  const float inv = 1.f/sum;
  #pragma unroll
  for (int p = 0; p < NP_; ++p) AW[(size_t)m*48 + h*NP_ + p] = e[p]*inv;
}

DEV float vcorner(const __bf16* vb, int xi, int yi, int d){
  if ((unsigned)xi >= (unsigned)WBEV_ || (unsigned)yi >= (unsigned)HBEV_) return 0.f;
  return (float)vb[((size_t)(yi*WBEV_ + xi))*DMODEL + d];
}

// deformable bilinear sampling: Y0[b,l,d] = sum_p aw * bilinear(v)
__global__ void sample_kernel(const __bf16* __restrict__ V, const float* __restrict__ REF,
                              const float* __restrict__ OA, const float* __restrict__ AW,
                              float* __restrict__ Y0){
  const int idx = blockIdx.x*256 + threadIdx.x;
  if (idx >= BATCH*LQ*DMODEL) return;
  const int d = idx & 127;
  const int l = (idx >> 7) % LQ;
  const int b = idx / (LQ*DMODEL);
  const int head = d >> 4;
  const int m = b*(HRV_*WRV_) + l/NQ_;     // unique query row
  const float* rp = REF + ((size_t)b*LQ + l)*2;
  const float rx = rp[0]*128.f - 0.5f, ry = rp[1]*128.f - 0.5f;
  const __bf16* vb = V + (size_t)b*PBEV*DMODEL;
  const float* op = OA + (size_t)m*144 + head*(NP_*2);
  const float* ap = AW + (size_t)m*48  + head*NP_;
  float acc = 0.f;
  #pragma unroll
  for (int p = 0; p < NP_; ++p){
    const float x = rx + op[p*2];
    const float y = ry + op[p*2+1];
    const float xf = floorf(x), yf = floorf(y);
    const float wx = x - xf, wy = y - yf;
    const int x0 = (int)xf, y0 = (int)yf;
    const float v00 = vcorner(vb, x0,   y0,   d);
    const float v01 = vcorner(vb, x0+1, y0,   d);
    const float v10 = vcorner(vb, x0,   y0+1, d);
    const float v11 = vcorner(vb, x0+1, y0+1, d);
    const float sx0 = v00 + (v01-v00)*wx;
    const float sx1 = v10 + (v11-v10)*wx;
    acc += ap[p]*(sx0 + (sx1-sx0)*wy);
  }
  Y0[idx] = acc;
}

// per-row layernorm over 256 channels
__global__ void layernorm_kernel(const float* __restrict__ x, const float* __restrict__ g,
                                 const float* __restrict__ bt, float* __restrict__ y){
  __shared__ float s1[256], s2[256];
  const int row = blockIdx.x, t = threadIdx.x;
  const float v = x[(size_t)row*256 + t];
  s1[t] = v; s2[t] = v*v; __syncthreads();
  for (int o = 128; o > 0; o >>= 1){
    if (t < o){ s1[t] += s1[t+o]; s2[t] += s2[t+o]; } __syncthreads();
  }
  const float mean = s1[0]*(1.f/256.f);
  const float inv  = rsqrtf(s2[0]*(1.f/256.f) - mean*mean + 1e-5f);
  y[(size_t)row*256 + t] = (v-mean)*inv*g[t] + bt[t];
}

// ---------------------------------------------------------------------------
// host-side NT dispatch for the strip GEMM
// ---------------------------------------------------------------------------
template<int EPI, bool ABF>
static void launch_nt(int NT, dim3 g, hipStream_t s, const void* A, const __bf16* W,
                      const float* bias, float* C, __bf16* Cb, const float* extra,
                      int emod, int M, int N, int K){
  if (NT == 4)      gemm_wmma<EPI,ABF,4><<<g,dim3(256),0,s>>>(A,W,bias,C,Cb,extra,emod,M,N,K);
  else if (NT == 3) gemm_wmma<EPI,ABF,3><<<g,dim3(256),0,s>>>(A,W,bias,C,Cb,extra,emod,M,N,K);
  else              gemm_wmma<EPI,ABF,1><<<g,dim3(256),0,s>>>(A,W,bias,C,Cb,extra,emod,M,N,K);
}

// ---------------------------------------------------------------------------
extern "C" void kernel_launch(void* const* d_in, const int* in_sizes, int n_in,
                              void* d_out, int out_size, void* d_ws, size_t ws_size,
                              hipStream_t stream){
  (void)in_sizes; (void)n_in; (void)out_size; (void)ws_size;
  const float* x_rv = (const float*)d_in[0];
  const float* bev  = (const float*)d_in[1];
  const float* l2e  = (const float*)d_in[2];
  const float* wq   = (const float*)d_in[3];
  const float* bq   = (const float*)d_in[4];
  const float* wv   = (const float*)d_in[5];
  const float* bv   = (const float*)d_in[6];
  const float* rw1  = (const float*)d_in[7];
  const float* rb1  = (const float*)d_in[8];
  const float* rg1  = (const float*)d_in[9];
  const float* rbt1 = (const float*)d_in[10];
  const float* rw2  = (const float*)d_in[11];
  const float* rg2  = (const float*)d_in[12];
  const float* rbt2 = (const float*)d_in[13];
  const float* rw3  = (const float*)d_in[14];
  const float* rb3  = (const float*)d_in[15];
  const float* mow  = (const float*)d_in[16];
  const float* mob  = (const float*)d_in[17];
  const float* maw  = (const float*)d_in[18];
  const float* mab  = (const float*)d_in[19];
  const float* mvw  = (const float*)d_in[20];
  const float* mvb  = (const float*)d_in[21];
  const float* mpw  = (const float*)d_in[22];
  const float* mpb  = (const float*)d_in[23];
  const float* ow1  = (const float*)d_in[24];
  const float* ob1  = (const float*)d_in[25];
  const float* ow2  = (const float*)d_in[26];
  const float* ob2  = (const float*)d_in[27];
  const float* ng1  = (const float*)d_in[28];
  const float* nb1  = (const float*)d_in[29];
  const float* fw1  = (const float*)d_in[30];
  const float* fb1  = (const float*)d_in[31];
  const float* fw2  = (const float*)d_in[32];
  const float* fb2  = (const float*)d_in[33];
  const float* ng2  = (const float*)d_in[34];
  const float* nb2  = (const float*)d_in[35];
  float* out = (float*)d_out;

  char* ws = (char*)d_ws;
  size_t off = 0;
  auto alloc = [&](size_t bytes)->char*{
    char* p = ws + off; off += (bytes + 255) & ~(size_t)255; return p;
  };
  __bf16* BEVT = (__bf16*)alloc(2ull*BATCH*PBEV*CBEV_);   // 134 MB
  __bf16* VV   = (__bf16*)alloc(2ull*BATCH*PBEV*DMODEL);  //  67 MB
  float*  PEt  = (float*) alloc(4ull*PBEV*DMODEL);
  float*  PEV  = (float*) alloc(4ull*PBEV*DMODEL);
  float*  Q0   = (float*) alloc(4ull*MRV*DMODEL);
  float*  OA   = (float*) alloc(4ull*MRV*144);
  float*  AW   = (float*) alloc(4ull*MRV*48);
  float*  REF  = (float*) alloc(4ull*BATCH*LQ*2);
  float*  UX   = (float*) alloc(4ull*128*NQ_);
  float*  UY   = (float*) alloc(4ull*128*NQ_);
  float*  UZ   = (float*) alloc(4ull*128*NQ_);
  float*  UD   = (float*) alloc(4ull*3*128);
  float*  H1RV = (float*) alloc(4ull*BATCH*128*128);
  float*  H2RV = (float*) alloc(4ull*BATCH*64*128);
  float*  Y0   = (float*) alloc(4ull*BATCH*LQ*DMODEL);    // 21 MB
  float*  YCK  = (float*) alloc(4ull*MRV*(DMODEL*NQ_));   // 21 MB
  float*  H1   = (float*) alloc(4ull*MRV*512);
  float*  Y2   = (float*) alloc(4ull*MRV*256);
  float*  YC   = (float*) alloc(4ull*MRV*256);
  float*  HF   = (float*) alloc(4ull*MRV*256);
  float*  YF   = (float*) alloc(4ull*MRV*256);
  float*  BIASV= (float*) alloc(4ull*128);
  float*  BOB  = (float*) alloc(4ull*144);
  __bf16* wqB  = (__bf16*)alloc(2ull*128*256);
  __bf16* wvT  = (__bf16*)alloc(2ull*256*128);
  __bf16* mvwB = (__bf16*)alloc(2ull*128*128);
  __bf16* WOA  = (__bf16*)alloc(2ull*144*128);
  __bf16* mpwB = (__bf16*)alloc(2ull*128*128);
  __bf16* ow1B = (__bf16*)alloc(2ull*512*2560);
  __bf16* ow2B = (__bf16*)alloc(2ull*256*512);
  __bf16* fw1B = (__bf16*)alloc(2ull*256*256);
  __bf16* fw2B = (__bf16*)alloc(2ull*256*256);
  __bf16* WCMB = (__bf16*)alloc(2ull*128*256);

  auto gemm = [&](int EPI, bool abf, const void* A, const __bf16* W, const float* bias,
                  float* C, __bf16* Cb, const float* extra, int emod, int M, int N, int K){
    const int tilesN = N >> 4;
    const int NT = (tilesN % 4 == 0) ? 4 : (tilesN % 3 == 0) ? 3 : 1;
    const int strips = tilesN / NT;
    dim3 g((unsigned)(((M >> 4)*strips + 7) / 8));
    if (abf){
      if (EPI == 2) launch_nt<2,true >(NT,g,stream,A,W,bias,C,Cb,extra,emod,M,N,K);
      else          launch_nt<0,true >(NT,g,stream,A,W,bias,C,Cb,extra,emod,M,N,K);
    } else {
      switch (EPI){
        case 0:  launch_nt<0,false>(NT,g,stream,A,W,bias,C,Cb,extra,emod,M,N,K); break;
        case 1:  launch_nt<1,false>(NT,g,stream,A,W,bias,C,Cb,extra,emod,M,N,K); break;
        case 2:  launch_nt<2,false>(NT,g,stream,A,W,bias,C,Cb,extra,emod,M,N,K); break;
        case 3:  launch_nt<3,false>(NT,g,stream,A,W,bias,C,Cb,extra,emod,M,N,K); break;
        default: launch_nt<4,false>(NT,g,stream,A,W,bias,C,Cb,extra,emod,M,N,K); break;
      }
    }
  };
  auto cvt = [&](const float* s, __bf16* d, int n){
    cvt_bf16_kernel<<<(n+255)/256, 256, 0, stream>>>(s, d, n);
  };

  // --- weights -> bf16 (once per call) ---
  cvt(wq,  wqB,  128*256);
  cvt_t_bf16_kernel<<<(128*256+255)/256,256,0,stream>>>(wv, wvT, 128, 256);
  cvt(mvw, mvwB, 128*128);
  cvt(mow, WOA,          96*128);
  cvt(maw, WOA + 96*128, 48*128);
  cvt(mpw, mpwB, 128*128);
  cvt(ow1, ow1B, 512*2560);
  cvt(ow2, ow2B, 256*512);
  cvt(fw1, fw1B, 256*256);
  cvt(fw2, fw2B, 256*256);
  copy_f32_kernel<<<1,256,0,stream>>>(mob, BOB,     96);
  copy_f32_kernel<<<1,256,0,stream>>>(mab, BOB+96,  48);

  // --- constants ---
  geom_kernel<<<1,128,0,stream>>>(UX, UY, UZ, UD);
  pe_kernel<<<(PBEV*DMODEL+255)/256,256,0,stream>>>(PEt);
  biasv_kernel<<<1,128,0,stream>>>(mvw, bv, mvb, BIASV);

  // --- BEV value path (fully fused: v = (mvw@wv)@bev + (mvw@PE + mvw@bv + mvb)) ---
  bev_transpose<<<BATCH*(CBEV_/32)*(PBEV/32),256,0,stream>>>(bev, BEVT);
  gemm(0,false, mvw, wvT, nullptr, nullptr, WCMB, nullptr, 1, 128, 256, 128);     // Wcomb
  gemm(0,false, PEt, mvwB, BIASV, PEV, nullptr, nullptr, 1, PBEV, 128, 128);      // PEV
  gemm(2,true,  BEVT, WCMB, nullptr, nullptr, VV, PEV, PBEV,
       BATCH*PBEV, 128, 256);                                                     // v (bf16)

  // --- query + sampling parameters ---
  gemm(0,false, x_rv, wqB, bq, Q0, nullptr, nullptr, 1, MRV, 128, 256);           // Q0
  gemm(0,false, Q0, WOA, BOB, OA, nullptr, nullptr, 1, MRV, 144, 128);            // off|aw
  softmax_aw_kernel<<<(MRV*NH_+255)/256,256,0,stream>>>(OA, AW);

  // --- range-view depth branch -> reference points ---
  rin_conv_kernel<<<(BATCH*128*128+255)/256,256,0,stream>>>(x_rv, UD, rw1, rb1, H1RV);
  gn_gelu_kernel<<<BATCH*8,256,0,stream>>>(H1RV, rg1, rbt1, 128, 16);
  conv3x3_kernel<<<(BATCH*64*128+255)/256,256,0,stream>>>(H1RV, rw2, H2RV);
  gn_gelu_kernel<<<BATCH*8,256,0,stream>>>(H2RV, rg2, rbt2, 64, 8);
  depth_ref_kernel<<<(BATCH*128+255)/256,256,0,stream>>>(H2RV, rw3, rb3, UX, UY, UZ, l2e, REF);

  // --- MSDA sampling + output projection (scatter into (d*NQ+q) channels) ---
  sample_kernel<<<(BATCH*LQ*DMODEL)/256,256,0,stream>>>(VV, REF, OA, AW, Y0);
  gemm(3,false, Y0, mpwB, mpb, YCK, nullptr, Q0, NQ_, BATCH*LQ, 128, 128);

  // --- output MLP + residual + LN + FFN + LN ---
  gemm(1,false, YCK, ow1B, ob1, H1, nullptr, nullptr, 1, MRV, 512, DMODEL*NQ_);
  gemm(4,false, H1,  ow2B, ob2, Y2, nullptr, x_rv,    1, MRV, 256, 512);
  layernorm_kernel<<<MRV,256,0,stream>>>(Y2, ng1, nb1, YC);
  gemm(1,false, YC,  fw1B, fb1, HF, nullptr, nullptr, 1, MRV, 256, 256);
  gemm(4,false, HF,  fw2B, fb2, YF, nullptr, YC,      1, MRV, 256, 256);
  layernorm_kernel<<<MRV,256,0,stream>>>(YF, ng2, nb2, out);
}
